// GCNGraph_5111011083136
// MI455X (gfx1250) — matrix operand order repaired and checked by reading, not compiled, not run on hardware
//
#include <hip/hip_runtime.h>
#include <math.h>

typedef __attribute__((ext_vector_type(16))) _Float16 v16h;
typedef __attribute__((ext_vector_type(8)))  _Float16 v8h;
typedef __attribute__((ext_vector_type(8)))  float    v8f;

#define HF 256   // hidden width
#define NB 16    // batch
#define NBLK 4   // N-dimension register blocking (4 x 16 = 64 cols per wave)

// ---------------- WMMA fragment helpers (wave32, f16 16x16x32 -> f32) -------
// ISA 7.12.2 layouts. All operands are pre-staged in f16 in the layout the
// fragment wants, so every fragment load is two 16B vector loads per lane.

// A (MxK) row-major f16: lane gets rows m0+(lane&15); two contiguous 8-half
// runs at k0+{0,16} (+8 for lanes>=16).
__device__ __forceinline__ v16h load_a_row16(const _Float16* __restrict__ p, int ld,
                                             int m0, int k0, int lane) {
  const _Float16* r = p + (long)(m0 + (lane & 15)) * ld + k0 + ((lane >= 16) ? 8 : 0);
  v8h lo = *(const v8h*)r;
  v8h hi = *(const v8h*)(r + 16);
  v16h a;
#pragma unroll
  for (int i = 0; i < 8; ++i) { a[i] = lo[i]; a[i + 8] = hi[i]; }
  return a;
}

// B (KxN) stored column-major (i.e. BT[n][k]): two contiguous 8-half runs at k0+{0,16} (+8 for lanes>=16).
__device__ __forceinline__ v16h load_b_cm16(const _Float16* __restrict__ p, int ld,
                                            int k0, int n0, int lane) {
  const _Float16* r = p + (long)(n0 + (lane & 15)) * ld + k0 + ((lane >= 16) ? 8 : 0);
  v8h lo = *(const v8h*)r;
  v8h hi = *(const v8h*)(r + 16);
  v16h b;
#pragma unroll
  for (int i = 0; i < 8; ++i) { b[i] = lo[i]; b[i + 8] = hi[i]; }
  return b;
}

__device__ __forceinline__ void store_c(float* __restrict__ p, int ld, int m0,
                                        int n0, int lane, v8f c) {
#pragma unroll
  for (int r = 0; r < 8; ++r) {
    int m = m0 + r + ((lane >= 16) ? 8 : 0);
    p[(long)m * ld + n0 + (lane & 15)] = c[r];
  }
}

// row-major f16 C store (wave-coalesced b16 stores)
__device__ __forceinline__ void store_c16row(_Float16* __restrict__ p, int ld, int m0,
                                             int n0, int lane, v8f c) {
#pragma unroll
  for (int r = 0; r < 8; ++r) {
    int m = m0 + r + ((lane >= 16) ? 8 : 0);
    p[(long)m * ld + n0 + (lane & 15)] = (_Float16)c[r];
  }
}

// transposed f16 C store: per lane 8 contiguous halves -> one b128 store
__device__ __forceinline__ void store_c16T(_Float16* __restrict__ p, int ldT, int m0,
                                           int n0, int lane, v8f c) {
  v8h h;
#pragma unroll
  for (int r = 0; r < 8; ++r) h[r] = (_Float16)c[r];
  *(v8h*)(p + (long)(n0 + (lane & 15)) * ldT + m0 + ((lane >= 16) ? 8 : 0)) = h;
}

#define WMMA(a, b, c) \
  __builtin_amdgcn_wmma_f32_16x16x32_f16(false, (a), false, (b), (short)0, (c), false, false)

// ---------------- WMMA GEMM kernels (16M x 64N tile per wave) ---------------
// C[b,m,ff] = sum_k AT[b][m][k] * BT[b][ff][k]  (agg = A^T h ; x_new = S^T h)
__global__ void k_gemm_tA16(const _Float16* __restrict__ AT, const _Float16* __restrict__ BT,
                            float* __restrict__ C, _Float16* __restrict__ C16,
                            int n, int ncols) {
  int b = blockIdx.z, m0 = blockIdx.y * 16, n0 = blockIdx.x * (16 * NBLK), lane = threadIdx.x;
  const _Float16* Ab = AT + (long)b * n * n;
  const _Float16* Bb = BT + (long)b * ncols * n;
  v8f c[NBLK] = {};
  for (int k0 = 0; k0 < n; k0 += 32) {
    if (k0 + 32 < n) {
      __builtin_prefetch(Ab + (long)(m0 + (lane & 15)) * n + k0 + 32, 0, 0);
      __builtin_prefetch(Bb + (long)(n0 + (lane & 31)) * n + k0 + 32, 0, 0);
    }
    v16h a = load_a_row16(Ab, n, m0, k0, lane);
#pragma unroll
    for (int j = 0; j < NBLK; ++j) {
      v16h bb = load_b_cm16(Bb, n, k0, n0 + 16 * j, lane);
      c[j] = WMMA(a, bb, c[j]);
    }
  }
  float* Cb = C + (long)b * n * ncols;
  _Float16* Cb16 = C16 + (long)b * n * ncols;
#pragma unroll
  for (int j = 0; j < NBLK; ++j) {
    store_c(Cb, ncols, m0, n0 + 16 * j, lane, c[j]);
    store_c16row(Cb16, ncols, m0, n0 + 16 * j, lane, c[j]);
  }
}

// g = relu(agg@Wrel^T + h@Wroot^T + brel) -> g f32 [n][HF] + gT16 [HF][n]
__global__ void k_gcn_lin16(const _Float16* __restrict__ agg16, const _Float16* __restrict__ h16,
                            const _Float16* __restrict__ Wrel, const _Float16* __restrict__ Wroot,
                            const float* __restrict__ brel, float* __restrict__ g,
                            _Float16* __restrict__ gT16, int n, int fin) {
  int b = blockIdx.z, m0 = blockIdx.y * 16, n0 = blockIdx.x * (16 * NBLK), lane = threadIdx.x;
  const _Float16* ab = agg16 + (long)b * n * fin;
  const _Float16* hb = h16 + (long)b * n * fin;
  v8f c[NBLK] = {};
  for (int k0 = 0; k0 < fin; k0 += 32) {
    v16h a0 = load_a_row16(ab, fin, m0, k0, lane);
    v16h a1 = load_a_row16(hb, fin, m0, k0, lane);
#pragma unroll
    for (int j = 0; j < NBLK; ++j) {
      v16h w0 = load_b_cm16(Wrel, fin, k0, n0 + 16 * j, lane);
      c[j] = WMMA(a0, w0, c[j]);
      v16h w1 = load_b_cm16(Wroot, fin, k0, n0 + 16 * j, lane);
      c[j] = WMMA(a1, w1, c[j]);
    }
  }
  float* gb = g + (long)b * n * HF;
  _Float16* gtb = gT16 + (long)b * HF * n;
#pragma unroll
  for (int j = 0; j < NBLK; ++j) {
    float bo = brel[n0 + 16 * j + (lane & 15)];
#pragma unroll
    for (int r = 0; r < 8; ++r) { float v = c[j][r] + bo; c[j][r] = v > 0.f ? v : 0.f; }
    store_c(gb, HF, m0, n0 + 16 * j, lane, c[j]);
    store_c16T(gtb, n, m0, n0 + 16 * j, lane, c[j]);
  }
}

// out = X @ W^T + bias (f32 out only; x_q linear, fin == HF)
__global__ void k_lin16(const _Float16* __restrict__ X16, const _Float16* __restrict__ W,
                        const float* __restrict__ bias, float* __restrict__ out, int n) {
  int b = blockIdx.z, m0 = blockIdx.y * 16, n0 = blockIdx.x * (16 * NBLK), lane = threadIdx.x;
  const _Float16* Xb = X16 + (long)b * n * HF;
  v8f c[NBLK] = {};
  for (int k0 = 0; k0 < HF; k0 += 32) {
    v16h a = load_a_row16(Xb, HF, m0, k0, lane);
#pragma unroll
    for (int j = 0; j < NBLK; ++j) {
      v16h w = load_b_cm16(W, HF, k0, n0 + 16 * j, lane);
      c[j] = WMMA(a, w, c[j]);
    }
  }
  float* ob = out + (long)b * n * HF;
#pragma unroll
  for (int j = 0; j < NBLK; ++j) {
    float bo = bias[n0 + 16 * j + (lane & 15)];
#pragma unroll
    for (int r = 0; r < 8; ++r) c[j][r] += bo;
    store_c(ob, HF, m0, n0 + 16 * j, lane, c[j]);
  }
}

// T1[b,s,j] = sum_t Asl[b][s][t] * SselT[b][j][t]; stored transposed f16
__global__ void k_T1_16(const _Float16* __restrict__ Asl16, const _Float16* __restrict__ SselT16,
                        _Float16* __restrict__ T1T16, int n, int kdim) {
  int b = blockIdx.z, m0 = blockIdx.y * 16, n0 = blockIdx.x * (16 * NBLK), lane = threadIdx.x;
  const _Float16* Ab = Asl16 + (long)b * n * n;
  const _Float16* Sb = SselT16 + (long)b * kdim * n;
  v8f c[NBLK] = {};
  for (int k0 = 0; k0 < n; k0 += 32) {
    v16h a = load_a_row16(Ab, n, m0, k0, lane);
#pragma unroll
    for (int j = 0; j < NBLK; ++j) {
      v16h bb = load_b_cm16(Sb, n, k0, n0 + 16 * j, lane);
      c[j] = WMMA(a, bb, c[j]);
    }
  }
  _Float16* Tb = T1T16 + (long)b * kdim * n;
#pragma unroll
  for (int j = 0; j < NBLK; ++j) store_c16T(Tb, n, m0, n0 + 16 * j, lane, c[j]);
}

// A'[b,i,j] = sum_s SselT[b][i][s] * T1T[b][j][s], diag removed.
// Emits next layer's three adjacency forms: f32, transposed f16, (+I) row f16.
__global__ void k_Anew16(const _Float16* __restrict__ SselT16, const _Float16* __restrict__ T1T16,
                         float* __restrict__ Anf, _Float16* __restrict__ AnT16,
                         _Float16* __restrict__ Ansl16, int n, int kdim) {
  int b = blockIdx.z, m0 = blockIdx.y * 16, n0 = blockIdx.x * (16 * NBLK), lane = threadIdx.x;
  const _Float16* Sb = SselT16 + (long)b * kdim * n;
  const _Float16* Tb = T1T16 + (long)b * kdim * n;
  v8f c[NBLK] = {};
  for (int k0 = 0; k0 < n; k0 += 32) {
    v16h a = load_a_row16(Sb, n, m0, k0, lane);
#pragma unroll
    for (int j = 0; j < NBLK; ++j) {
      v16h bb = load_b_cm16(Tb, n, k0, n0 + 16 * j, lane);
      c[j] = WMMA(a, bb, c[j]);
    }
  }
  float* Af = Anf + (long)b * kdim * kdim;
  _Float16* At = AnT16 + (long)b * kdim * kdim;
  _Float16* As = Ansl16 + (long)b * kdim * kdim;
#pragma unroll
  for (int j = 0; j < NBLK; ++j) {
    int ncol = n0 + 16 * j + (lane & 15);
#pragma unroll
    for (int r = 0; r < 8; ++r) {
      int m = m0 + r + ((lane >= 16) ? 8 : 0);
      if (m == ncol) c[j][r] = 0.f;
      Af[(long)m * kdim + ncol] = c[j][r];
      As[(long)m * kdim + ncol] = (_Float16)(c[j][r] + ((m == ncol) ? 1.f : 0.f));
    }
    store_c16T(At, kdim, m0, n0 + 16 * j, lane, c[j]);
  }
}

// ---------------- layout prep kernels ---------------------------------------
// adjacency: AT16[b][t][s] = adj[b][s][t]; Asl16[b][s][t] = adj + I
__global__ void k_prep_adj(const float* __restrict__ adj, _Float16* __restrict__ AT16,
                           _Float16* __restrict__ Asl16, int n) {
  int b = blockIdx.y, s = blockIdx.x;
  const float* ar = adj + ((long)b * n + s) * n;
  for (int t = threadIdx.x; t < n; t += blockDim.x) {
    float v = ar[t];
    Asl16[((long)b * n + s) * n + t] = (_Float16)(v + ((s == t) ? 1.f : 0.f));
    AT16[((long)b * n + t) * n + s] = (_Float16)v;
  }
}

// x: x16 row-major f16 + xT16 transposed f16
__global__ void k_prep_x(const float* __restrict__ x, _Float16* __restrict__ x16,
                         _Float16* __restrict__ xT16, int n, int f) {
  int b = blockIdx.y, s = blockIdx.x;
  const float* xr = x + ((long)b * n + s) * f;
  for (int ff = threadIdx.x; ff < f; ff += blockDim.x) {
    _Float16 v = (_Float16)xr[ff];
    x16[((long)b * n + s) * f + ff] = v;
    xT16[((long)b * f + ff) * n + s] = v;
  }
}

// flat f32 -> f16 (weights)
__global__ void k_cvt(const float* __restrict__ src, _Float16* __restrict__ dst, int count) {
  int i = blockIdx.x * blockDim.x + threadIdx.x;
  if (i < count) dst[i] = (_Float16)src[i];
}

// ---------------- VALU kernels ----------------------------------------------
// x_qmax16[b,t,f] = max over in-neighbors s (A_sl mask) of g[b,s,f]  (f16 out)
__global__ void k_xqmax(const float* __restrict__ A, const float* __restrict__ g,
                        _Float16* __restrict__ xq16, int n) {
  int b = blockIdx.y, t = blockIdx.x;
  __shared__ unsigned char msk[256];
  for (int s = threadIdx.x; s < n; s += blockDim.x)
    msk[s] = (A[((long)b * n + s) * n + t] > 0.f) || (s == t);
  __syncthreads();
  const float* gb = g + (long)b * n * HF;
  for (int ff = threadIdx.x; ff < HF; ff += blockDim.x) {
    float m = -INFINITY;
    for (int s = 0; s < n; ++s)
      if (msk[s]) { float v = gb[(long)s * HF + ff]; m = v > m ? v : m; }
    xq16[((long)b * n + t) * HF + ff] = (_Float16)m;
  }
}

// p = g . att_w[H:], q = x_q . att_w[:H]  (one wave per node)
__global__ void k_pq(const float* __restrict__ g, const float* __restrict__ xq2,
                     const float* __restrict__ att_w, float* __restrict__ p,
                     float* __restrict__ q, int n) {
  int b = blockIdx.y, node = blockIdx.x, lane = threadIdx.x;
  const float* gr = g + ((long)b * n + node) * HF;
  const float* xr = xq2 + ((long)b * n + node) * HF;
  float sp = 0.f, sq = 0.f;
  for (int f = lane; f < HF; f += 32) { sp += gr[f] * att_w[HF + f]; sq += xr[f] * att_w[f]; }
  for (int o = 16; o > 0; o >>= 1) { sp += __shfl_down(sp, o, 32); sq += __shfl_down(sq, o, 32); }
  if (lane == 0) { p[b * n + node] = sp; q[b * n + node] = sq; }
}

// masked softmax over sources s per dst t; writes S f32 and S^T f16
__global__ void k_softmax(const float* __restrict__ A, const float* __restrict__ p,
                          const float* __restrict__ q, const float* __restrict__ attb,
                          float* __restrict__ S, _Float16* __restrict__ ST16, int n) {
  int b = blockIdx.y, t = blockIdx.x, s = threadIdx.x;
  __shared__ float red[256];
  bool valid = s < n;
  bool msk = false;
  float raw = -INFINITY;
  if (valid) {
    msk = (A[((long)b * n + s) * n + t] > 0.f) || (s == t);
    if (msk) {
      float r = p[b * n + s] + q[b * n + t] + attb[0];
      raw = r > 0.f ? r : 0.2f * r;
    }
  }
  red[s] = raw; __syncthreads();
  for (int st = 128; st > 0; st >>= 1) {
    if (s < st) red[s] = fmaxf(red[s], red[s + st]);
    __syncthreads();
  }
  float mx = red[0]; __syncthreads();
  float e = (valid && msk) ? __expf(raw - mx) : 0.f;
  red[s] = e; __syncthreads();
  for (int st = 128; st > 0; st >>= 1) {
    if (s < st) red[s] += red[s + st];
    __syncthreads();
  }
  if (valid) {
    float v = e / red[0];
    S[((long)b * n + s) * n + t] = v;
    ST16[((long)b * n + t) * n + s] = (_Float16)v;   // coalesced in s
  }
}

// LEConv per-node dots: a1 = x.w1+b1, v2 = x.w2, v3 = x.w3 (one wave per node)
__global__ void k_ledots(const float* __restrict__ xnew, const float* __restrict__ w1,
                         const float* __restrict__ b1p, const float* __restrict__ w2,
                         const float* __restrict__ w3, float* __restrict__ a1,
                         float* __restrict__ v2, float* __restrict__ v3, int n) {
  int b = blockIdx.y, node = blockIdx.x, lane = threadIdx.x;
  const float* xr = xnew + ((long)b * n + node) * HF;
  float s1 = 0.f, s2 = 0.f, s3 = 0.f;
  for (int f = lane; f < HF; f += 32) { float v = xr[f]; s1 += v * w1[f]; s2 += v * w2[f]; s3 += v * w3[f]; }
  for (int o = 16; o > 0; o >>= 1) {
    s1 += __shfl_down(s1, o, 32); s2 += __shfl_down(s2, o, 32); s3 += __shfl_down(s3, o, 32);
  }
  if (lane == 0) { a1[b * n + node] = s1 + b1p[0]; v2[b * n + node] = s2; v3[b * n + node] = s3; }
}

// fit[b,t] = sigmoid(sum_s mf*a1[s] - deg*v2[t] + v3[t] + b3)
__global__ void k_fit(const float* __restrict__ A, const float* __restrict__ a1,
                      const float* __restrict__ v2, const float* __restrict__ v3,
                      const float* __restrict__ b3p, float* __restrict__ fit, int n) {
  int b = blockIdx.y, t = blockIdx.x, s = threadIdx.x;
  __shared__ float rs[256], rd[256];
  float acc = 0.f, deg = 0.f;
  if (s < n) {
    bool msk = (A[((long)b * n + s) * n + t] > 0.f) || (s == t);
    if (msk) { acc = a1[b * n + s]; deg = 1.f; }
  }
  rs[s] = acc; rd[s] = deg; __syncthreads();
  for (int st = 128; st > 0; st >>= 1) {
    if (s < st) { rs[s] += rs[s + st]; rd[s] += rd[s + st]; }
    __syncthreads();
  }
  if (s == 0) {
    float z = rs[0] - rd[0] * v2[b * n + t] + v3[b * n + t] + b3p[0];
    fit[b * n + t] = 1.f / (1.f + __expf(-z));
  }
}

// exact top-k by rank (unique via index tie-break) -> idx sorted descending
__global__ void k_topk(const float* __restrict__ fit, int* __restrict__ idx, int n, int k) {
  int b = blockIdx.x, t = threadIdx.x;
  __shared__ float f[256];
  if (t < n) f[t] = fit[b * n + t];
  __syncthreads();
  if (t < n) {
    float ft = f[t];
    int rank = 0;
    for (int s = 0; s < n; ++s) { float fs = f[s]; rank += (fs > ft) || (fs == ft && s < t); }
    if (rank < k) idx[b * k + rank] = t;
  }
}

// xsel (f32 for readout) + xsel16 row + xselT16 col for next layer's GEMMs
__global__ void k_xsel(const float* __restrict__ xnew, const float* __restrict__ fit,
                       const int* __restrict__ idx, float* __restrict__ xsel,
                       _Float16* __restrict__ xsel16, _Float16* __restrict__ xselT16,
                       int n, int k) {
  int b = blockIdx.y, j = blockIdx.x;
  int src = idx[b * k + j];
  float fv = fit[b * n + src];
  const float* xr = xnew + ((long)b * n + src) * HF;
  float* xo = xsel + ((long)b * k + j) * HF;
  _Float16* xo16 = xsel16 + ((long)b * k + j) * HF;
  for (int f = threadIdx.x; f < HF; f += blockDim.x) {
    float v = xr[f] * fv;
    xo[f] = v;
    xo16[f] = (_Float16)v;
    xselT16[((long)b * HF + f) * k + j] = (_Float16)v;
  }
}

// SselT16[b][j][s] = S[b][s][idx[j]]  (f16, column-major staging)
__global__ void k_sselT(const float* __restrict__ S, const int* __restrict__ idx,
                        _Float16* __restrict__ SselT, int n, int k) {
  int b = blockIdx.y, j = blockIdx.x;
  int t = idx[b * k + j];
  for (int s = threadIdx.x; s < n; s += blockDim.x)
    SselT[((long)b * k + j) * n + s] = (_Float16)S[((long)b * n + s) * n + t];
}

// ro[b, 0:H] (+)= mean_j xsel, ro[b, H:2H] (+)= max_j xsel
__global__ void k_readout(const float* __restrict__ xsel, float* __restrict__ ro,
                          int k, int first) {
  int b = blockIdx.x, f = threadIdx.x;   // blockDim == HF
  const float* xb = xsel + (long)b * k * HF;
  float sm = 0.f, mx = -INFINITY;
  for (int j = 0; j < k; ++j) { float v = xb[(long)j * HF + f]; sm += v; mx = v > mx ? v : mx; }
  float mean = sm / (float)k;
  if (first) { ro[b * 2 * HF + f] = mean; ro[b * 2 * HF + HF + f] = mx; }
  else       { ro[b * 2 * HF + f] += mean; ro[b * 2 * HF + HF + f] += mx; }
}

// out = relu(ro @ W1^T + b1) @ W2^T + b2    (W1:[256,512], W2:[2,256])
__global__ void k_mlp(const float* __restrict__ ro, const float* __restrict__ W1,
                      const float* __restrict__ b1, const float* __restrict__ W2,
                      const float* __restrict__ b2, float* __restrict__ out) {
  int b = blockIdx.x, o = threadIdx.x;   // 256 threads
  __shared__ float hid[256];
  const float* r = ro + b * 2 * HF;
  float acc = b1[o];
  for (int i = 0; i < 2 * HF; ++i) acc += r[i] * W1[o * 2 * HF + i];
  hid[o] = acc > 0.f ? acc : 0.f;
  __syncthreads();
  if (o < 2) {
    float a = b2[o];
    for (int i = 0; i < HF; ++i) a += hid[i] * W2[o * HF + i];
    out[b * 2 + o] = a;
  }
}

// ---------------- host driver -----------------------------------------------
extern "C" void kernel_launch(void* const* d_in, const int* in_sizes, int n_in,
                              void* d_out, int out_size, void* d_ws, size_t ws_size,
                              hipStream_t stream) {
  (void)in_sizes; (void)n_in; (void)out_size; (void)ws_size;
  const float* x     = (const float*)d_in[0];
  const float* adj   = (const float*)d_in[1];
  const float* Wrel[3]  = {(const float*)d_in[2], (const float*)d_in[3], (const float*)d_in[4]};
  const float* brel[3]  = {(const float*)d_in[5], (const float*)d_in[6], (const float*)d_in[7]};
  const float* Wroot[3] = {(const float*)d_in[8], (const float*)d_in[9], (const float*)d_in[10]};
  const float* Wq    = (const float*)d_in[11];
  const float* bq    = (const float*)d_in[12];
  const float* att_w = (const float*)d_in[13];
  const float* att_b = (const float*)d_in[14];
  const float* w1    = (const float*)d_in[15];
  const float* b1    = (const float*)d_in[16];
  const float* w2    = (const float*)d_in[17];
  const float* w3    = (const float*)d_in[18];
  const float* b3    = (const float*)d_in[19];
  const float* W_l1  = (const float*)d_in[20];
  const float* b_l1  = (const float*)d_in[21];
  const float* W_l2  = (const float*)d_in[22];
  const float* b_l2  = (const float*)d_in[23];
  float* out = (float*)d_out;

  char* ws = (char*)d_ws;
  size_t off = 0;
  auto carve = [&](size_t bytes) -> void* {
    void* pp = (void*)(ws + off);
    off += (bytes + 255) & ~(size_t)255;
    return pp;
  };
  // f32 buffers
  float* agg   = (float*)carve((size_t)NB * 256 * 256 * 4);
  float* g     = (float*)carve((size_t)NB * 256 * 256 * 4);
  float* xq2   = (float*)carve((size_t)NB * 256 * 256 * 4);
  float* S     = (float*)carve((size_t)NB * 256 * 256 * 4);
  float* xnew  = (float*)carve((size_t)NB * 256 * 256 * 4);
  float* xsel  = (float*)carve((size_t)NB * 128 * 256 * 4);
  float* A0f   = (float*)carve((size_t)NB * 128 * 128 * 4);
  float* A1f   = (float*)carve((size_t)NB * 64 * 64 * 4);
  float* pv    = (float*)carve((size_t)NB * 256 * 4);
  float* qv    = (float*)carve((size_t)NB * 256 * 4);
  float* a1v   = (float*)carve((size_t)NB * 256 * 4);
  float* v2v   = (float*)carve((size_t)NB * 256 * 4);
  float* v3v   = (float*)carve((size_t)NB * 256 * 4);
  float* fitv  = (float*)carve((size_t)NB * 256 * 4);
  int*   idx   = (int*)carve((size_t)NB * 128 * 4);
  float* ro    = (float*)carve((size_t)NB * 512 * 4);
  // f16 staging buffers
  _Float16* adjT16  = (_Float16*)carve((size_t)NB * 256 * 256 * 2);
  _Float16* adjSl16 = (_Float16*)carve((size_t)NB * 256 * 256 * 2);
  _Float16* A1T16   = (_Float16*)carve((size_t)NB * 128 * 128 * 2);
  _Float16* A1sl16  = (_Float16*)carve((size_t)NB * 128 * 128 * 2);
  _Float16* A2T16   = (_Float16*)carve((size_t)NB * 64 * 64 * 2);
  _Float16* A2sl16  = (_Float16*)carve((size_t)NB * 64 * 64 * 2);
  _Float16* x16     = (_Float16*)carve((size_t)NB * 256 * 128 * 2);
  _Float16* xT16    = (_Float16*)carve((size_t)NB * 256 * 128 * 2);
  _Float16* xsel16  = (_Float16*)carve((size_t)NB * 128 * 256 * 2);
  _Float16* xselT16 = (_Float16*)carve((size_t)NB * 256 * 128 * 2);
  _Float16* agg16   = (_Float16*)carve((size_t)NB * 256 * 256 * 2);
  _Float16* gT16    = (_Float16*)carve((size_t)NB * 256 * 256 * 2);
  _Float16* ST16    = (_Float16*)carve((size_t)NB * 256 * 256 * 2);
  _Float16* xnew16  = (_Float16*)carve((size_t)NB * 256 * 256 * 2);
  _Float16* xqmax16 = (_Float16*)carve((size_t)NB * 256 * 256 * 2);
  _Float16* SselT16 = (_Float16*)carve((size_t)NB * 128 * 256 * 2);
  _Float16* T1T16   = (_Float16*)carve((size_t)NB * 128 * 256 * 2);
  _Float16* Wrel16[3], *Wroot16[3];
  Wrel16[0]  = (_Float16*)carve((size_t)256 * 128 * 2);
  Wrel16[1]  = (_Float16*)carve((size_t)256 * 256 * 2);
  Wrel16[2]  = (_Float16*)carve((size_t)256 * 256 * 2);
  Wroot16[0] = (_Float16*)carve((size_t)256 * 128 * 2);
  Wroot16[1] = (_Float16*)carve((size_t)256 * 256 * 2);
  Wroot16[2] = (_Float16*)carve((size_t)256 * 256 * 2);
  _Float16* Wq16 = (_Float16*)carve((size_t)256 * 256 * 2);

  // weight conversions (once per call)
  k_cvt<<<dim3(256 * 128 / 256), 256, 0, stream>>>(Wrel[0], Wrel16[0], 256 * 128);
  k_cvt<<<dim3(256 * 256 / 256), 256, 0, stream>>>(Wrel[1], Wrel16[1], 256 * 256);
  k_cvt<<<dim3(256 * 256 / 256), 256, 0, stream>>>(Wrel[2], Wrel16[2], 256 * 256);
  k_cvt<<<dim3(256 * 128 / 256), 256, 0, stream>>>(Wroot[0], Wroot16[0], 256 * 128);
  k_cvt<<<dim3(256 * 256 / 256), 256, 0, stream>>>(Wroot[1], Wroot16[1], 256 * 256);
  k_cvt<<<dim3(256 * 256 / 256), 256, 0, stream>>>(Wroot[2], Wroot16[2], 256 * 256);
  k_cvt<<<dim3(256 * 256 / 256), 256, 0, stream>>>(Wq, Wq16, 256 * 256);
  k_prep_adj<<<dim3(256, NB), 256, 0, stream>>>(adj, adjT16, adjSl16, 256);
  k_prep_x<<<dim3(256, NB), 128, 0, stream>>>(x, x16, xT16, 256, 128);

  const int ns[3]   = {256, 128, 64};
  const int fins[3] = {128, 256, 256};
  float*     Afn[2]  = {A0f, A1f};
  _Float16*  ATn[2]  = {A1T16, A2T16};
  _Float16*  ASLn[2] = {A1sl16, A2sl16};

  const float* Acur = adj;            // f32 adjacency (mask checks)
  const _Float16* ATcur = adjT16;     // transposed f16 (A^T h GEMM)
  const _Float16* ASLcur = adjSl16;   // (+I) row-major f16 (coarsening GEMM)
  const _Float16* h16 = x16;          // row-major f16 features
  const _Float16* hT16 = xT16;        // transposed f16 features
  dim3 wave(32);
  for (int l = 0; l < 3; ++l) {
    int n = ns[l], fin = fins[l], kk = n / 2;
    k_gemm_tA16<<<dim3(fin / 64, n / 16, NB), wave, 0, stream>>>(ATcur, hT16, agg, agg16, n, fin);
    k_gcn_lin16<<<dim3(HF / 64, n / 16, NB), wave, 0, stream>>>(agg16, h16, Wrel16[l], Wroot16[l], brel[l], g, gT16, n, fin);
    k_xqmax<<<dim3(n, NB), 256, 0, stream>>>(Acur, g, xqmax16, n);
    k_lin16<<<dim3(HF / 64, n / 16, NB), wave, 0, stream>>>(xqmax16, Wq16, bq, xq2, n);
    k_pq<<<dim3(n, NB), 32, 0, stream>>>(g, xq2, att_w, pv, qv, n);
    k_softmax<<<dim3(n, NB), 256, 0, stream>>>(Acur, pv, qv, att_b, S, ST16, n);
    k_gemm_tA16<<<dim3(HF / 64, n / 16, NB), wave, 0, stream>>>(ST16, gT16, xnew, xnew16, n, HF);
    k_ledots<<<dim3(n, NB), 32, 0, stream>>>(xnew, w1, b1, w2, w3, a1v, v2v, v3v, n);
    k_fit<<<dim3(n, NB), 256, 0, stream>>>(Acur, a1v, v2v, v3v, b3, fitv, n);
    k_topk<<<NB, 256, 0, stream>>>(fitv, idx, n, kk);
    k_xsel<<<dim3(kk, NB), 256, 0, stream>>>(xnew, fitv, idx, xsel, xsel16, xselT16, n, kk);
    k_readout<<<NB, 256, 0, stream>>>(xsel, ro, kk, l == 0 ? 1 : 0);
    if (l < 2) {  // last layer's coarsened adjacency is never used
      k_sselT<<<dim3(kk, NB), 256, 0, stream>>>(S, idx, SselT16, n, kk);
      k_T1_16<<<dim3(kk / 64, n / 16, NB), wave, 0, stream>>>(ASLcur, SselT16, T1T16, n, kk);
      k_Anew16<<<dim3(kk / 64, kk / 16, NB), wave, 0, stream>>>(SselT16, T1T16, Afn[l], ATn[l], ASLn[l], n, kk);
      Acur = Afn[l]; ATcur = ATn[l]; ASLcur = ASLn[l];
    }
    h16 = xsel16; hT16 = xselT16;
  }
  k_mlp<<<NB, 256, 0, stream>>>(ro, W_l1, b_l1, W_l2, b_l2, out);
}
